// SAE_NTM_16604343566988
// MI455X (gfx1250) — compile-verified, matrix operand order checked by reading, not run on hardware
//
#include <hip/hip_runtime.h>
#include <math.h>

// ---------------------------------------------------------------------------
// NTM forward loss on MI455X (gfx1250). Heavy GEMMs via v_wmma_f32_16x16x32_f16
// (f16 in, f32 accumulate). Compile-time NV (N-edge) and MT (M-tiles per wave)
// keep the WMMA loop branch-free: no exec-masked WMMA, no phi copies. MT=2
// (32x64 C tile per wave) is used for the high-parallelism GEMMs (k/v: 6144
// waves, decoder: 5008 waves) cutting fragment loads per WMMA from 2.5 to 1.5;
// MT=1 elsewhere to preserve occupancy (M=512 GEMMs have few M-tiles).
// ---------------------------------------------------------------------------

typedef __attribute__((ext_vector_type(16))) _Float16 v16h;
typedef __attribute__((ext_vector_type(8)))  _Float16 v8h;
typedef __attribute__((ext_vector_type(8)))  float    v8f;

#define B_    512
#define M_    64
#define V_    20000
#define E_    1024
#define D_    384
#define T_    100
#define TP_   112     // T padded to multiple of 16 for WMMA N-tiles
#define KDEC_ 128     // dec K (=T) padded to multiple of 32 for WMMA K-steps
#define EPS_  1e-5f

union HFrag { v16h v; v8h h[2]; };

__device__ __forceinline__ float softplusf(float x) {
  return x > 20.f ? x : log1pf(expf(x));
}

// ---------------- f32 -> f16 convert with zero padding ----------------------
__global__ void cvt_pad_kernel(const float* __restrict__ src, _Float16* __restrict__ dst,
                               int n, int k, int npad, int kpad) {
  size_t total = (size_t)npad * kpad;
  for (size_t i = (size_t)blockIdx.x * blockDim.x + threadIdx.x; i < total;
       i += (size_t)gridDim.x * blockDim.x) {
    int r = (int)(i / kpad), c = (int)(i % kpad);
    dst[i] = (r < n && c < k) ? (_Float16)src[(size_t)r * k + c] : (_Float16)0.f;
  }
}

// ---------------- WMMA GEMM: C = epi(A @ W^T + bias) ------------------------
// A: [M,K] f16 row-major. W: [N,K] f16 row-major (W row n == B column n;
// identical per-lane fragment layout for A and B per CDNA5 ISA 7.12.2).
// Wave tile: MT*16(M) x NV*16(N), all compile-time -> branch-free WMMA loop
// (EXEC all-ones). Requires M % (16*MT) == 0, K % 32 == 0.
template <int EPI, int NV, int MT>
__global__ void __launch_bounds__(256) gemm_wmma_kernel(
    const _Float16* __restrict__ A, const _Float16* __restrict__ Wm,
    const float* __restrict__ bias, int nbias,
    float* __restrict__ Cf, _Float16* __restrict__ Ch,
    int M, int K, int ldc, int ntBase, int groupsN) {
  const int wavesM = M / (16 * MT);
  const int wid = blockIdx.x * 8 + (threadIdx.x >> 5);
  if (wid >= wavesM * groupsN) return;                 // whole-wave exit only

  const int tm = wid / groupsN, tg = wid % groupsN;
  const int m0 = tm * (16 * MT);
  const int nt0 = ntBase + (tg << 2);

  const int lane = threadIdx.x & 31;
  const int r = lane & 15, h = lane >> 4;              // row-in-tile, K-half

  const _Float16* arow[MT];
#pragma unroll
  for (int mi = 0; mi < MT; ++mi)
    arow[mi] = A + (size_t)(m0 + mi * 16 + r) * K + h * 8;
  const _Float16* wrow[NV];
#pragma unroll
  for (int j = 0; j < NV; ++j)
    wrow[j] = Wm + (size_t)(((nt0 + j) << 4) + r) * K + h * 8;

  v8f acc[MT][NV];
#pragma unroll
  for (int mi = 0; mi < MT; ++mi)
#pragma unroll
    for (int j = 0; j < NV; ++j)
#pragma unroll
      for (int i = 0; i < 8; ++i) acc[mi][j][i] = 0.f;

  int k0 = 0;
  // K unrolled x2: all loads of both steps issue as clauses, then 2*MT*NV
  // back-to-back v_wmma -> load latency of step i+1 hides under WMMA of i.
  for (; k0 + 64 <= K; k0 += 64) {
    if (k0 + 192 < K) __builtin_prefetch(arow[0] + k0 + 192);  // global_prefetch_b8
    HFrag fa0[MT], fa1[MT], fb0[NV], fb1[NV];
#pragma unroll
    for (int mi = 0; mi < MT; ++mi) {                          // ISA 16-bit layout:
      fa0[mi].h[0] = *(const v8h*)(arow[mi] + k0);             // K=h*8..+7
      fa0[mi].h[1] = *(const v8h*)(arow[mi] + k0 + 16);        // K=16+h*8..+7
      fa1[mi].h[0] = *(const v8h*)(arow[mi] + k0 + 32);
      fa1[mi].h[1] = *(const v8h*)(arow[mi] + k0 + 48);
    }
#pragma unroll
    for (int j = 0; j < NV; ++j) {
      fb0[j].h[0] = *(const v8h*)(wrow[j] + k0);
      fb0[j].h[1] = *(const v8h*)(wrow[j] + k0 + 16);
      fb1[j].h[0] = *(const v8h*)(wrow[j] + k0 + 32);
      fb1[j].h[1] = *(const v8h*)(wrow[j] + k0 + 48);
    }
#pragma unroll
    for (int mi = 0; mi < MT; ++mi)
#pragma unroll
      for (int j = 0; j < NV; ++j)
        acc[mi][j] = __builtin_amdgcn_wmma_f32_16x16x32_f16(
            false, fa0[mi].v, false, fb0[j].v, (short)0, acc[mi][j], false, false);
#pragma unroll
    for (int mi = 0; mi < MT; ++mi)
#pragma unroll
      for (int j = 0; j < NV; ++j)
        acc[mi][j] = __builtin_amdgcn_wmma_f32_16x16x32_f16(
            false, fa1[mi].v, false, fb1[j].v, (short)0, acc[mi][j], false, false);
  }
  if (k0 < K) {                                                // K%64==32 tail
    HFrag fa[MT], fb[NV];
#pragma unroll
    for (int mi = 0; mi < MT; ++mi) {
      fa[mi].h[0] = *(const v8h*)(arow[mi] + k0);
      fa[mi].h[1] = *(const v8h*)(arow[mi] + k0 + 16);
    }
#pragma unroll
    for (int j = 0; j < NV; ++j) {
      fb[j].h[0] = *(const v8h*)(wrow[j] + k0);
      fb[j].h[1] = *(const v8h*)(wrow[j] + k0 + 16);
    }
#pragma unroll
    for (int mi = 0; mi < MT; ++mi)
#pragma unroll
      for (int j = 0; j < NV; ++j)
        acc[mi][j] = __builtin_amdgcn_wmma_f32_16x16x32_f16(
            false, fa[mi].v, false, fb[j].v, (short)0, acc[mi][j], false, false);
  }

#pragma unroll
  for (int mi = 0; mi < MT; ++mi)
#pragma unroll
    for (int j = 0; j < NV; ++j) {
      const int col = ((nt0 + j) << 4) + r;
      const float bb = (col < nbias) ? bias[col] : 0.f;
#pragma unroll
      for (int i = 0; i < 8; ++i) {                            // C layout: VGPR i ->
        const int row = m0 + mi * 16 + h * 8 + i;              // M=h*8+i, N=lane&15
        float v = acc[mi][j][i] + bb;
        if (EPI == 1) v = softplusf(v);
        if (Cf) Cf[(size_t)row * ldc + col] = v;
        if (Ch) Ch[(size_t)row * ldc + col] = (_Float16)v;
      }
    }
}

// ---------------- ragged single-query attention -----------------------------
__global__ void attn_kernel(const float* __restrict__ q, const _Float16* __restrict__ kh,
                            const _Float16* __restrict__ vh, const int* __restrict__ lengths,
                            float* __restrict__ di, _Float16* __restrict__ dih) {
  const int b = blockIdx.x, t = threadIdx.x;          // 256 threads
  __shared__ float qs[D_];
  __shared__ float sc[M_];
  __shared__ float al[M_];
  for (int i = t; i < D_; i += 256) qs[i] = q[(size_t)b * D_ + i];
  __syncthreads();

  const int wave = t >> 5, lane = t & 31;
  const float scale = rsqrtf((float)D_);
  for (int mi = 0; mi < 8; ++mi) {
    const int m = wave * 8 + mi;
    const _Float16* krow = kh + ((size_t)b * M_ + m) * D_;
    float p = 0.f;
    for (int d = lane; d < D_; d += 32) p += qs[d] * (float)krow[d];
    for (int off = 16; off > 0; off >>= 1) p += __shfl_xor(p, off, 32);
    if (lane == 0) sc[m] = p * scale;
  }
  __syncthreads();

  if (t == 0) {
    int len = lengths[b]; if (len < 1) len = 1; if (len > M_) len = M_;
    float mx = -INFINITY;
    for (int m = 0; m < len; ++m) mx = fmaxf(mx, sc[m]);
    float s = 0.f;
    for (int m = 0; m < M_; ++m) { float e = (m < len) ? expf(sc[m] - mx) : 0.f; al[m] = e; s += e; }
    const float inv = 1.f / s;
    for (int m = 0; m < M_; ++m) al[m] *= inv;
  }
  __syncthreads();

  for (int d = t; d < D_; d += 256) {
    float acc = 0.f;
    for (int m = 0; m < M_; ++m) acc += al[m] * (float)vh[((size_t)b * M_ + m) * D_ + d];
    di[(size_t)b * D_ + d] = acc;
    dih[(size_t)b * D_ + d] = (_Float16)acc;
  }
}

// ---------------- BatchNorm1d (training batch stats, weight=1) --------------
// One block per column; biased variance; in-place safe (column-exclusive).
__global__ void bn_kernel(const float* __restrict__ in, int rows, int ldin,
                          const float* __restrict__ bias,
                          float* __restrict__ out, int ldout) {
  const int c = blockIdx.x, t = threadIdx.x;          // 256 threads
  __shared__ float s1[256], s2[256];
  float a = 0.f, b = 0.f;
  for (int r = t; r < rows; r += 256) { float v = in[(size_t)r * ldin + c]; a += v; b += v * v; }
  s1[t] = a; s2[t] = b; __syncthreads();
  for (int s = 128; s > 0; s >>= 1) {
    if (t < s) { s1[t] += s1[t + s]; s2[t] += s2[t + s]; }
    __syncthreads();
  }
  const float mean = s1[0] / rows;
  const float var  = s2[0] / rows - mean * mean;
  const float istd = rsqrtf(var + EPS_);
  const float bb = bias[c];
  for (int r = t; r < rows; r += 256)
    out[(size_t)r * ldout + c] = (in[(size_t)r * ldin + c] - mean) * istd + bb;
}

// ---------------- theta = softmax(mu) (padded f16) + per-row KLD ------------
__global__ void theta_kld_kernel(const float* __restrict__ mu, const float* __restrict__ lv,
                                 _Float16* __restrict__ theta_h, float* __restrict__ kld) {
  const int b = blockIdx.x, t = threadIdx.x;          // 128 threads (== KDEC_)
  __shared__ float sv[128];
  __shared__ float smx, ssum;
  const float m = (t < T_) ? mu[(size_t)b * T_ + t] : -INFINITY;
  sv[t] = m; __syncthreads();
  if (t == 0) { float mx = -INFINITY; for (int i = 0; i < T_; ++i) mx = fmaxf(mx, sv[i]); smx = mx; }
  __syncthreads();
  const float e = (t < T_) ? expf(m - smx) : 0.f;
  sv[t] = e; __syncthreads();
  if (t == 0) { float s = 0.f; for (int i = 0; i < T_; ++i) s += sv[i]; ssum = s; }
  __syncthreads();
  theta_h[(size_t)b * KDEC_ + t] = (_Float16)(e / ssum);

  const float var2 = 1.f - 1.f / (float)T_;
  float term = 0.f;
  if (t < T_) {
    const float l = lv[(size_t)b * T_ + t];
    const float var = expf(l);
    term = var / var2 + m * m / var2 + logf(var2) - l;
  }
  sv[t] = term; __syncthreads();
  if (t == 0) { float s = 0.f; for (int i = 0; i < T_; ++i) s += sv[i]; kld[b] = 0.5f * (s - (float)T_); }
}

// ---------------- per-row recon loss: -(x * log_softmax(lb)).sum ------------
__global__ void recon_kernel(const float* __restrict__ lb, const float* __restrict__ x,
                             float* __restrict__ recon) {
  const int b = blockIdx.x, t = threadIdx.x;          // 256 threads
  __shared__ float red[256];
  const float* row = lb + (size_t)b * V_;
  const float* xr  = x  + (size_t)b * V_;
  float mx = -INFINITY;
  for (int v = t; v < V_; v += 256) mx = fmaxf(mx, row[v]);
  red[t] = mx; __syncthreads();
  for (int s = 128; s > 0; s >>= 1) { if (t < s) red[t] = fmaxf(red[t], red[t + s]); __syncthreads(); }
  mx = red[0]; __syncthreads();
  float se = 0.f;
  for (int v = t; v < V_; v += 256) se += expf(row[v] - mx);
  red[t] = se; __syncthreads();
  for (int s = 128; s > 0; s >>= 1) { if (t < s) red[t] += red[t + s]; __syncthreads(); }
  const float lse = mx + logf(red[0]); __syncthreads();
  float acc = 0.f;
  for (int v = t; v < V_; v += 256) acc += xr[v] * (row[v] - lse);
  red[t] = acc; __syncthreads();
  for (int s = 128; s > 0; s >>= 1) { if (t < s) red[t] += red[t + s]; __syncthreads(); }
  if (t == 0) recon[b] = -red[0];
}

// ---------------- final scalar: mean(recon) + mean(kld) ---------------------
__global__ void final_kernel(const float* __restrict__ recon, const float* __restrict__ kld,
                             float* __restrict__ out) {
  const int t = threadIdx.x;                          // 256 threads
  __shared__ float r1[256], r2[256];
  float a = 0.f, b = 0.f;
  for (int i = t; i < B_; i += 256) { a += recon[i]; b += kld[i]; }
  r1[t] = a; r2[t] = b; __syncthreads();
  for (int s = 128; s > 0; s >>= 1) {
    if (t < s) { r1[t] += r1[t + s]; r2[t] += r2[t + s]; }
    __syncthreads();
  }
  if (t == 0) out[0] = r1[0] / (float)B_ + r2[0] / (float)B_;
}

// ---------------------------------------------------------------------------
static inline int cvt_grid(size_t total) {
  size_t blk = (total + 255) / 256;
  return (int)(blk > 16384 ? 16384 : blk);
}

// Main launch (NV=4 full groups) + compile-time remainder launch (NV=1..3).
template <int EPI, int MT>
static void launch_gemm_epi(const _Float16* A, const _Float16* W,
                            const float* bias, int nbias, float* Cf, _Float16* Ch,
                            int M, int N, int K, int ldc, hipStream_t s) {
  const int wavesM = M / (16 * MT);
  const int tilesN = N / 16;
  const int fullG  = tilesN / 4;
  const int rem    = tilesN - fullG * 4;
  if (fullG > 0) {
    const int blocks = (wavesM * fullG + 7) / 8;
    gemm_wmma_kernel<EPI, 4, MT><<<blocks, 256, 0, s>>>(A, W, bias, nbias, Cf, Ch,
                                                        M, K, ldc, 0, fullG);
  }
  if (rem > 0) {
    const int blocks = (wavesM + 7) / 8;
    switch (rem) {
      case 1: gemm_wmma_kernel<EPI, 1, MT><<<blocks, 256, 0, s>>>(A, W, bias, nbias, Cf, Ch,
                                                                  M, K, ldc, fullG * 4, 1); break;
      case 2: gemm_wmma_kernel<EPI, 2, MT><<<blocks, 256, 0, s>>>(A, W, bias, nbias, Cf, Ch,
                                                                  M, K, ldc, fullG * 4, 1); break;
      default: gemm_wmma_kernel<EPI, 3, MT><<<blocks, 256, 0, s>>>(A, W, bias, nbias, Cf, Ch,
                                                                   M, K, ldc, fullG * 4, 1); break;
    }
  }
}

// mt=2 only for high-parallelism GEMMs (requires M%32==0); mt=1 preserves
// occupancy for the small-M (512-row) GEMMs.
static void launch_gemm(int epi, int mt, const _Float16* A, const _Float16* W,
                        const float* bias, int nbias, float* Cf, _Float16* Ch,
                        int M, int N, int K, int ldc, hipStream_t s) {
  if (epi == 1) {
    if (mt == 2) launch_gemm_epi<1, 2>(A, W, bias, nbias, Cf, Ch, M, N, K, ldc, s);
    else         launch_gemm_epi<1, 1>(A, W, bias, nbias, Cf, Ch, M, N, K, ldc, s);
  } else {
    if (mt == 2) launch_gemm_epi<0, 2>(A, W, bias, nbias, Cf, Ch, M, N, K, ldc, s);
    else         launch_gemm_epi<0, 1>(A, W, bias, nbias, Cf, Ch, M, N, K, ldc, s);
  }
}

extern "C" void kernel_launch(void* const* d_in, const int* in_sizes, int n_in,
                              void* d_out, int out_size, void* d_ws, size_t ws_size,
                              hipStream_t stream) {
  (void)in_sizes; (void)n_in; (void)out_size; (void)ws_size;

  const float* x         = (const float*)d_in[0];
  const float* emb       = (const float*)d_in[1];
  const int*   lengths   = (const int*)  d_in[2];
  const float* fc11_w    = (const float*)d_in[3];
  const float* fc11_b    = (const float*)d_in[4];
  const float* fc12_w    = (const float*)d_in[5];
  const float* fc12_b    = (const float*)d_in[6];
  const float* q_w       = (const float*)d_in[7];
  const float* q_b       = (const float*)d_in[8];
  const float* k_w       = (const float*)d_in[9];
  const float* k_b       = (const float*)d_in[10];
  const float* v_w       = (const float*)d_in[11];
  const float* v_b       = (const float*)d_in[12];
  const float* mean_w    = (const float*)d_in[13];
  const float* mean_b    = (const float*)d_in[14];
  const float* logvar_w  = (const float*)d_in[15];
  const float* logvar_b  = (const float*)d_in[16];
  const float* mean_bn_b = (const float*)d_in[17];
  const float* lv_bn_b   = (const float*)d_in[18];
  const float* dec_w     = (const float*)d_in[19];
  const float* dec_b     = (const float*)d_in[20];
  const float* dec_bn_b  = (const float*)d_in[21];

  // ---- workspace carve (256B aligned); total ~192 MB -----------------------
  char* p = (char*)d_ws;
  auto alloc = [&](size_t bytes) -> char* {
    char* r = p; p += (bytes + 255) & ~(size_t)255; return r;
  };
  _Float16* x_h     = (_Float16*)alloc((size_t)B_ * V_ * 2);
  _Float16* fc11_h  = (_Float16*)alloc((size_t)E_ * V_ * 2);
  _Float16* fc12_h  = (_Float16*)alloc((size_t)E_ * E_ * 2);
  _Float16* qw_h    = (_Float16*)alloc((size_t)D_ * E_ * 2);
  _Float16* kw_h    = (_Float16*)alloc((size_t)D_ * D_ * 2);
  _Float16* vw_h    = (_Float16*)alloc((size_t)D_ * D_ * 2);
  _Float16* emb_h   = (_Float16*)alloc((size_t)B_ * M_ * D_ * 2);
  _Float16* meanw_h = (_Float16*)alloc((size_t)TP_ * D_ * 2);
  _Float16* lvw_h   = (_Float16*)alloc((size_t)TP_ * D_ * 2);
  _Float16* decw_h  = (_Float16*)alloc((size_t)V_ * KDEC_ * 2);
  _Float16* e1_h    = (_Float16*)alloc((size_t)B_ * E_ * 2);
  _Float16* hbow_h  = (_Float16*)alloc((size_t)B_ * E_ * 2);
  float*    q_f     = (float*)   alloc((size_t)B_ * D_ * 4);
  _Float16* k_h     = (_Float16*)alloc((size_t)B_ * M_ * D_ * 2);
  _Float16* v_h     = (_Float16*)alloc((size_t)B_ * M_ * D_ * 2);
  float*    di_f    = (float*)   alloc((size_t)B_ * D_ * 4);
  _Float16* di_h    = (_Float16*)alloc((size_t)B_ * D_ * 2);
  float*    mu_pre  = (float*)   alloc((size_t)B_ * TP_ * 4);
  float*    lv_pre  = (float*)   alloc((size_t)B_ * TP_ * 4);
  float*    mu_bn   = (float*)   alloc((size_t)B_ * T_ * 4);
  float*    lv_bn   = (float*)   alloc((size_t)B_ * T_ * 4);
  _Float16* theta_h = (_Float16*)alloc((size_t)B_ * KDEC_ * 2);
  float*    logit   = (float*)   alloc((size_t)B_ * V_ * 4);
  float*    recon_v = (float*)   alloc((size_t)B_ * 4);
  float*    kld_v   = (float*)   alloc((size_t)B_ * 4);

  // ---- f32 -> f16 conversions (zero-padded where WMMA tiling needs it) -----
  cvt_pad_kernel<<<cvt_grid((size_t)B_*V_),       256, 0, stream>>>(x,       x_h,     B_,     V_, B_,     V_);
  cvt_pad_kernel<<<cvt_grid((size_t)E_*V_),       256, 0, stream>>>(fc11_w,  fc11_h,  E_,     V_, E_,     V_);
  cvt_pad_kernel<<<cvt_grid((size_t)E_*E_),       256, 0, stream>>>(fc12_w,  fc12_h,  E_,     E_, E_,     E_);
  cvt_pad_kernel<<<cvt_grid((size_t)D_*E_),       256, 0, stream>>>(q_w,     qw_h,    D_,     E_, D_,     E_);
  cvt_pad_kernel<<<cvt_grid((size_t)D_*D_),       256, 0, stream>>>(k_w,     kw_h,    D_,     D_, D_,     D_);
  cvt_pad_kernel<<<cvt_grid((size_t)D_*D_),       256, 0, stream>>>(v_w,     vw_h,    D_,     D_, D_,     D_);
  cvt_pad_kernel<<<cvt_grid((size_t)B_*M_*D_),    256, 0, stream>>>(emb,     emb_h,   B_*M_,  D_, B_*M_,  D_);
  cvt_pad_kernel<<<cvt_grid((size_t)TP_*D_),      256, 0, stream>>>(mean_w,  meanw_h, T_,     D_, TP_,    D_);
  cvt_pad_kernel<<<cvt_grid((size_t)TP_*D_),      256, 0, stream>>>(logvar_w,lvw_h,   T_,     D_, TP_,    D_);
  cvt_pad_kernel<<<cvt_grid((size_t)V_*KDEC_),    256, 0, stream>>>(dec_w,   decw_h,  V_,     T_, V_,     KDEC_);

  // ---- encoder MLP (MT=1: only 512 rows -> keep wave parallelism) ----------
  launch_gemm(1, 1, x_h,    fc11_h, fc11_b, E_, nullptr, e1_h,   B_, E_, V_, E_, stream); // softplus
  launch_gemm(1, 1, e1_h,   fc12_h, fc12_b, E_, nullptr, hbow_h, B_, E_, E_, E_, stream); // softplus
  launch_gemm(0, 1, hbow_h, qw_h,   q_b,    D_, q_f,     nullptr,B_, D_, E_, D_, stream);

  // ---- k/v projections over all (b,m) (MT=2: 6144 waves, B-frag reuse) -----
  launch_gemm(0, 2, emb_h, kw_h, k_b, D_, nullptr, k_h, B_*M_, D_, D_, D_, stream);
  launch_gemm(0, 2, emb_h, vw_h, v_b, D_, nullptr, v_h, B_*M_, D_, D_, D_, stream);

  // ---- ragged attention ----------------------------------------------------
  attn_kernel<<<B_, 256, 0, stream>>>(q_f, k_h, v_h, lengths, di_f, di_h);

  // ---- mean / logvar heads + batch norm ------------------------------------
  launch_gemm(0, 1, di_h, meanw_h, mean_b,   T_, mu_pre, nullptr, B_, TP_, D_, TP_, stream);
  launch_gemm(0, 1, di_h, lvw_h,   logvar_b, T_, lv_pre, nullptr, B_, TP_, D_, TP_, stream);
  bn_kernel<<<T_, 256, 0, stream>>>(mu_pre, B_, TP_, mean_bn_b, mu_bn, T_);
  bn_kernel<<<T_, 256, 0, stream>>>(lv_pre, B_, TP_, lv_bn_b,   lv_bn, T_);

  // ---- theta + KLD ---------------------------------------------------------
  theta_kld_kernel<<<B_, 128, 0, stream>>>(mu_bn, lv_bn, theta_h, kld_v);

  // ---- decoder (MT=2: 5008 waves) + BN + recon loss ------------------------
  launch_gemm(0, 2, theta_h, decw_h, dec_b, V_, logit, nullptr, B_, V_, KDEC_, V_, stream);
  bn_kernel<<<V_, 256, 0, stream>>>(logit, B_, V_, dec_bn_b, logit, V_);   // in-place
  recon_kernel<<<B_, 256, 0, stream>>>(logit, x, recon_v);

  // ---- final scalar --------------------------------------------------------
  final_kernel<<<1, 256, 0, stream>>>(recon_v, kld_v, (float*)d_out);
}
